// MambaLayer_29910152249588
// MI455X (gfx1250) — matrix operand order repaired and checked by reading, not compiled, not run on hardware
//
#include <hip/hip_runtime.h>
#include <hip/hip_bf16.h>
#include <math.h>

// ---------------------------------------------------------------------------
// Mamba layer for MI455X (gfx1250, wave32, WMMA).
//   DIM=128, D_STATE=16, D_CONV=4, D_INNER=256, DT_RANK=8, L=8192, BATCH=2
// ---------------------------------------------------------------------------

typedef __attribute__((ext_vector_type(16))) _Float16 v16h;
typedef __attribute__((ext_vector_type(8)))  float    v8f;

constexpr int BATCHN = 2;
constexpr int CH     = 128;   // DIM
constexpr int LSEQ   = 16 * 16 * 32;         // 8192
constexpr int MROWS  = BATCHN * LSEQ;        // 16384 token rows
constexpr int DI     = 256;   // D_INNER
constexpr int DSTATE = 16;
constexpr int DTR    = 8;     // DT_RANK
constexpr int NPROJ  = DTR + 2 * DSTATE;     // 40
constexpr int NPAD   = 48;    // padded to WMMA tile multiple

// ---- workspace layout (bytes) ----------------------------------------------
constexpr size_t OFF_XN16  = 0;
constexpr size_t SZ_XN16   = (size_t)MROWS * CH * 2;        // xn as f16
constexpr size_t OFF_WIN16 = OFF_XN16 + SZ_XN16;
constexpr size_t SZ_WIN16  = (size_t)(2 * DI) * CH * 2;     // in_proj_w f16
constexpr size_t OFF_XZ    = OFF_WIN16 + SZ_WIN16;
constexpr size_t SZ_XZ     = (size_t)MROWS * (2 * DI) * 4;  // xz f32
constexpr size_t OFF_XC    = OFF_XZ + SZ_XZ;
constexpr size_t SZ_XC     = (size_t)MROWS * DI * 4;        // conv out f32
constexpr size_t OFF_XC16  = OFF_XC + SZ_XC;
constexpr size_t SZ_XC16   = (size_t)MROWS * DI * 2;        // conv out f16
constexpr size_t OFF_WXP16 = OFF_XC16 + SZ_XC16;
constexpr size_t SZ_WXP16  = (size_t)NPAD * DI * 2;         // padded x_proj_w
constexpr size_t OFF_DBL   = OFF_WXP16 + SZ_WXP16;
constexpr size_t SZ_DBL    = (size_t)MROWS * NPAD * 4;      // dt|B|C f32
constexpr size_t OFF_DELTA = OFF_DBL + SZ_DBL;
constexpr size_t SZ_DELTA  = (size_t)MROWS * DI * 4;
constexpr size_t OFF_Y     = OFF_DELTA + SZ_DELTA;
constexpr size_t SZ_Y      = (size_t)MROWS * DI * 4;        // scan output
constexpr size_t OFF_Y16   = OFF_Y + SZ_Y;
constexpr size_t SZ_Y16    = (size_t)MROWS * DI * 2;        // gated y f16
constexpr size_t OFF_WOUT16 = OFF_Y16 + SZ_Y16;
constexpr size_t SZ_WOUT16 = (size_t)CH * DI * 2;           // out_proj_w f16

// ---------------------------------------------------------------------------
// WMMA fragment loaders (16-bit A 16x32 and B 32x16 lane layouts per ISA 7.12.2)
// ---------------------------------------------------------------------------
__device__ inline v16h load_A_frag(const _Float16* __restrict__ A, int lda,
                                   int row0, int k0, int lane) {
  const int m  = lane & 15;
  const int kO = (lane < 16) ? 0 : 8;
  const _Float16* p = A + (size_t)(row0 + m) * lda + k0 + kO;
  v16h a;
#pragma unroll
  for (int j = 0; j < 8; ++j)  a[j] = p[j];        // K = kO + 0..7
#pragma unroll
  for (int j = 8; j < 16; ++j) a[j] = p[j + 8];    // K = kO + 16..23
  return a;
}

// B(k,n) = W[n][k]   (W row-major, N x K)
__device__ inline v16h load_B_frag(const _Float16* __restrict__ W, int ldw,
                                   int col0, int k0, int lane) {
  const int n  = lane & 15;
  const int kO = (lane < 16) ? 0 : 8;
  const _Float16* p = W + (size_t)(col0 + n) * ldw + k0 + kO;
  v16h b;
#pragma unroll
  for (int j = 0; j < 8; ++j)  b[j] = p[j];        // K = kO + 0..7
#pragma unroll
  for (int j = 8; j < 16; ++j) b[j] = p[j + 8];    // K = kO + 16..23
  return b;
}

// ---------------------------------------------------------------------------
// 1) LayerNorm over C=128, one wave per token row, wave32 butterfly reduction
// ---------------------------------------------------------------------------
__global__ void ln_f16_kernel(const float* __restrict__ x,
                              const float* __restrict__ w,
                              const float* __restrict__ bsc,
                              _Float16* __restrict__ xn) {
  const int wid  = (int)((blockIdx.x * blockDim.x + threadIdx.x) >> 5);
  const int lane = (int)(threadIdx.x & 31);
  if (wid >= MROWS) return;
  const int b = wid / LSEQ;
  const int l = wid - b * LSEQ;
  const float* xb = x + ((size_t)b * CH) * LSEQ + l;

  float v[4];
  float s = 0.f;
#pragma unroll
  for (int j = 0; j < 4; ++j) {
    const int c = lane + 32 * j;
    v[j] = xb[(size_t)c * LSEQ];
    s += v[j];
  }
  for (int off = 16; off; off >>= 1) s += __shfl_xor(s, off, 32);
  const float mu = s * (1.f / 128.f);

  float q = 0.f;
#pragma unroll
  for (int j = 0; j < 4; ++j) { const float d = v[j] - mu; q += d * d; }
  for (int off = 16; off; off >>= 1) q += __shfl_xor(q, off, 32);
  const float inv = rsqrtf(q * (1.f / 128.f) + 1e-5f);

#pragma unroll
  for (int j = 0; j < 4; ++j) {
    const int c = lane + 32 * j;
    const float r = (v[j] - mu) * inv * w[c] + bsc[c];
    xn[(size_t)wid * CH + c] = (_Float16)r;
  }
}

// ---------------------------------------------------------------------------
// 2) weight conversions
// ---------------------------------------------------------------------------
__global__ void cvt_f16_kernel(const float* __restrict__ s,
                               _Float16* __restrict__ d, int n) {
  const int i = (int)(blockIdx.x * blockDim.x + threadIdx.x);
  if (i < n) d[i] = (_Float16)s[i];
}

__global__ void pad_xproj_kernel(const float* __restrict__ s,
                                 _Float16* __restrict__ d) {
  const int i = (int)(blockIdx.x * blockDim.x + threadIdx.x);
  if (i >= NPAD * DI) return;
  const int r = i / DI, c = i - r * DI;
  d[i] = (r < NPROJ) ? (_Float16)s[r * DI + c] : (_Float16)0.f;
}

// ---------------------------------------------------------------------------
// 3) N-blocked WMMA GEMM:  C[M,N] = A[M,K] * W[N,K]^T  (f16 x f16 -> f32)
//    Each wave computes one 16-row M tile and NB consecutive 16-col N tiles:
//    one A-fragment load feeds NB WMMAs (NB x A-traffic reduction).
// ---------------------------------------------------------------------------
template <int NB>
__global__ void wmma_gemm_nb_kernel(const _Float16* __restrict__ A, int lda,
                                    const _Float16* __restrict__ W, int ldw,
                                    float* __restrict__ Cmat, int ldc,
                                    int Mt, int NtB, int K) {
  const int wid  = (int)((blockIdx.x * blockDim.x + threadIdx.x) >> 5);
  const int lane = (int)(threadIdx.x & 31);
  if (wid >= Mt * NtB) return;               // whole wave exits: EXEC stays all-1
  const int tm  = wid / NtB, tnb = wid - tm * NtB;
  const int row0 = tm * 16;
  const int col0 = tnb * (NB * 16);

  v8f acc[NB];
#pragma unroll
  for (int j = 0; j < NB; ++j) acc[j] = (v8f){};

  for (int k0 = 0; k0 < K; k0 += 32) {
    const v16h a = load_A_frag(A, lda, row0, k0, lane);
#pragma unroll
    for (int j = 0; j < NB; ++j) {
      const v16h b = load_B_frag(W, ldw, col0 + j * 16, k0, lane);
      acc[j] = __builtin_amdgcn_wmma_f32_16x16x32_f16(false, a, false, b,
                                                      (short)0, acc[j],
                                                      false, false);
    }
  }

  const int n  = lane & 15;
  const int m0 = (lane < 16) ? 0 : 8;
#pragma unroll
  for (int j = 0; j < NB; ++j)
#pragma unroll
    for (int r = 0; r < 8; ++r)
      Cmat[(size_t)(row0 + m0 + r) * ldc + col0 + j * 16 + n] = acc[j][r];
}

// out_proj variant: store transposed straight into (B, C, L) output layout
template <int NB>
__global__ void wmma_gemm_out_kernel(const _Float16* __restrict__ A, int lda,
                                     const _Float16* __restrict__ W, int ldw,
                                     float* __restrict__ out,
                                     int Mt, int NtB, int K) {
  const int wid  = (int)((blockIdx.x * blockDim.x + threadIdx.x) >> 5);
  const int lane = (int)(threadIdx.x & 31);
  if (wid >= Mt * NtB) return;
  const int tm  = wid / NtB, tnb = wid - tm * NtB;
  const int row0 = tm * 16;
  const int col0 = tnb * (NB * 16);

  v8f acc[NB];
#pragma unroll
  for (int j = 0; j < NB; ++j) acc[j] = (v8f){};

  for (int k0 = 0; k0 < K; k0 += 32) {
    const v16h a = load_A_frag(A, lda, row0, k0, lane);
#pragma unroll
    for (int j = 0; j < NB; ++j) {
      const v16h b = load_B_frag(W, ldw, col0 + j * 16, k0, lane);
      acc[j] = __builtin_amdgcn_wmma_f32_16x16x32_f16(false, a, false, b,
                                                      (short)0, acc[j],
                                                      false, false);
    }
  }

  const int n  = lane & 15;
  const int m0 = (lane < 16) ? 0 : 8;
#pragma unroll
  for (int j = 0; j < NB; ++j) {
#pragma unroll
    for (int r = 0; r < 8; ++r) {
      const int mg = row0 + m0 + r;               // token row = b*L + l
      const int bb = mg / LSEQ;
      const int l  = mg - bb * LSEQ;
      out[((size_t)bb * CH + (col0 + j * 16 + n)) * LSEQ + l] = acc[j][r];
    }
  }
}

// ---------------------------------------------------------------------------
// 4) depthwise causal conv (d_conv=4) + SiLU, writes f32 + f16
// ---------------------------------------------------------------------------
__global__ void conv_silu_kernel(const float* __restrict__ xz,
                                 const float* __restrict__ cw,
                                 const float* __restrict__ cb,
                                 float* __restrict__ xc,
                                 _Float16* __restrict__ xc16) {
  const int i = (int)(blockIdx.x * blockDim.x + threadIdx.x);
  if (i >= MROWS * DI) return;
  const int d = i & (DI - 1);
  const int m = i >> 8;
  const int b = m / LSEQ;
  const int l = m - b * LSEQ;

  float acc = cb[d];
#pragma unroll
  for (int k = 0; k < 4; ++k) {
    const int lp = l - 3 + k;
    if (lp >= 0)
      acc += cw[d * 4 + k] * xz[((size_t)(b * LSEQ + lp)) * (2 * DI) + d];
  }
  const float sv = acc / (1.f + __expf(-acc));   // SiLU
  xc[i]   = sv;
  xc16[i] = (_Float16)sv;
}

// ---------------------------------------------------------------------------
// 5) delta = softplus(dt @ dt_proj_w^T + dt_proj_b)   (K = DT_RANK = 8)
// ---------------------------------------------------------------------------
__global__ void dt_softplus_kernel(const float* __restrict__ dbl,
                                   const float* __restrict__ dtw,
                                   const float* __restrict__ dtb,
                                   float* __restrict__ delta) {
  const int i = (int)(blockIdx.x * blockDim.x + threadIdx.x);
  if (i >= MROWS * DI) return;
  const int d = i & (DI - 1);
  const int m = i >> 8;
  const float* row = dbl + (size_t)m * NPAD;
  float s = dtb[d];
#pragma unroll
  for (int r = 0; r < DTR; ++r) s += row[r] * dtw[d * DTR + r];
  delta[i] = (s > 20.f) ? s : log1pf(__expf(s));
}

// ---------------------------------------------------------------------------
// 6) selective scan: h[d,n] register-resident over all 8192 steps.
//    Wave owns 2 channels; lanes = (half, n).  y = sum_n h*C via shfl_xor.
// ---------------------------------------------------------------------------
__global__ void scan_kernel(const float* __restrict__ delta,
                            const float* __restrict__ xc,
                            const float* __restrict__ dbl,
                            const float* __restrict__ A_log,
                            float* __restrict__ y) {
  const int lane = (int)(threadIdx.x & 31);
  const int wave = (int)(threadIdx.x >> 5);          // 0..7
  const int half = lane >> 4;
  const int n    = lane & 15;
  const int blk  = (int)blockIdx.x;                  // 32 blocks
  const int b    = blk >> 4;
  const int d    = (blk & 15) * 16 + wave * 2 + half;

  const float Av = -__expf(A_log[d * DSTATE + n]);
  float h = 0.f;
  const size_t mrow = (size_t)b * LSEQ;

  for (int l = 0; l < LSEQ; ++l) {
    const size_t m = mrow + l;
    if (l + 8 < LSEQ)
      __builtin_prefetch(dbl + (m + 8) * NPAD, 0, 1);
    const float dt = delta[m * DI + d];
    const float u  = xc[m * DI + d];
    const float* dr = dbl + m * NPAD;
    const float Bv = dr[DTR + n];
    const float Cv = dr[DTR + DSTATE + n];

    h = __expf(dt * Av) * h + dt * u * Bv;
    float p = h * Cv;
    p += __shfl_xor(p, 1, 32);
    p += __shfl_xor(p, 2, 32);
    p += __shfl_xor(p, 4, 32);
    p += __shfl_xor(p, 8, 32);
    if (n == 0) y[m * DI + d] = p;
  }
}

// ---------------------------------------------------------------------------
// 7) gate: y16 = (y + xc*D) * silu(z)
// ---------------------------------------------------------------------------
__global__ void combine_kernel(const float* __restrict__ y,
                               const float* __restrict__ xc,
                               const float* __restrict__ Dp,
                               const float* __restrict__ xz,
                               _Float16* __restrict__ y16) {
  const int i = (int)(blockIdx.x * blockDim.x + threadIdx.x);
  if (i >= MROWS * DI) return;
  const int d = i & (DI - 1);
  const size_t m = (size_t)(i >> 8);
  const float yv = y[i] + xc[i] * Dp[d];
  const float z  = xz[m * (2 * DI) + DI + d];
  const float sz = z / (1.f + __expf(-z));
  y16[i] = (_Float16)(yv * sz);
}

// ---------------------------------------------------------------------------
// launch
// ---------------------------------------------------------------------------
extern "C" void kernel_launch(void* const* d_in, const int* in_sizes, int n_in,
                              void* d_out, int out_size, void* d_ws, size_t ws_size,
                              hipStream_t stream) {
  (void)in_sizes; (void)n_in; (void)out_size; (void)ws_size;
  const float* x         = (const float*)d_in[0];
  const float* norm_w    = (const float*)d_in[1];
  const float* norm_b    = (const float*)d_in[2];
  const float* in_proj_w = (const float*)d_in[3];
  const float* conv_w    = (const float*)d_in[4];
  const float* conv_b    = (const float*)d_in[5];
  const float* x_proj_w  = (const float*)d_in[6];
  const float* dt_proj_w = (const float*)d_in[7];
  const float* dt_proj_b = (const float*)d_in[8];
  const float* A_log     = (const float*)d_in[9];
  const float* Dparam    = (const float*)d_in[10];
  const float* out_proj_w= (const float*)d_in[11];
  float* out = (float*)d_out;

  char* ws = (char*)d_ws;
  _Float16* xn16   = (_Float16*)(ws + OFF_XN16);
  _Float16* win16  = (_Float16*)(ws + OFF_WIN16);
  float*    xz     = (float*)   (ws + OFF_XZ);
  float*    xc     = (float*)   (ws + OFF_XC);
  _Float16* xc16   = (_Float16*)(ws + OFF_XC16);
  _Float16* wxp16  = (_Float16*)(ws + OFF_WXP16);
  float*    dbl    = (float*)   (ws + OFF_DBL);
  float*    delta  = (float*)   (ws + OFF_DELTA);
  float*    yscan  = (float*)   (ws + OFF_Y);
  _Float16* y16    = (_Float16*)(ws + OFF_Y16);
  _Float16* wout16 = (_Float16*)(ws + OFF_WOUT16);

  const int TB = 256;

  // 1) LayerNorm -> f16 (one wave per row: MROWS waves)
  ln_f16_kernel<<<(MROWS * 32 + TB - 1) / TB, TB, 0, stream>>>(x, norm_w, norm_b, xn16);

  // 2) weight conversions
  cvt_f16_kernel<<<((2 * DI * CH) + TB - 1) / TB, TB, 0, stream>>>(in_proj_w, win16, 2 * DI * CH);
  cvt_f16_kernel<<<((CH * DI) + TB - 1) / TB, TB, 0, stream>>>(out_proj_w, wout16, CH * DI);
  pad_xproj_kernel<<<((NPAD * DI) + TB - 1) / TB, TB, 0, stream>>>(x_proj_w, wxp16);

  // 3) in_proj: xz[M,512] = xn[M,128] @ in_proj_w[512,128]^T   (NB=4)
  {
    const int Mt = MROWS / 16;                   // 1024
    const int NtB = (2 * DI) / 16 / 4;           // 8 wave-columns of 4 tiles
    const int waves = Mt * NtB;                  // 8192
    wmma_gemm_nb_kernel<4><<<(waves * 32 + TB - 1) / TB, TB, 0, stream>>>(
        xn16, CH, win16, CH, xz, 2 * DI, Mt, NtB, CH);
  }

  // 4) conv + SiLU
  conv_silu_kernel<<<(MROWS * DI + TB - 1) / TB, TB, 0, stream>>>(xz, conv_w, conv_b, xc, xc16);

  // 5) x_proj: dbl[M,48] = xc[M,256] @ wxp[48,256]^T   (NB=3: one wave = all N)
  {
    const int Mt = MROWS / 16;                   // 1024
    const int NtB = 1;
    const int waves = Mt * NtB;                  // 1024
    wmma_gemm_nb_kernel<3><<<(waves * 32 + TB - 1) / TB, TB, 0, stream>>>(
        xc16, DI, wxp16, DI, dbl, NPAD, Mt, NtB, DI);
  }

  // 6) delta
  dt_softplus_kernel<<<(MROWS * DI + TB - 1) / TB, TB, 0, stream>>>(dbl, dt_proj_w, dt_proj_b, delta);

  // 7) selective scan (32 blocks x 256 thr: B*DI channels, register h state)
  scan_kernel<<<BATCHN * (DI / 16), TB, 0, stream>>>(delta, xc, dbl, A_log, yscan);

  // 8) gate
  combine_kernel<<<(MROWS * DI + TB - 1) / TB, TB, 0, stream>>>(yscan, xc, Dparam, xz, y16);

  // 9) out_proj with transposed store into (B, C, L)   (NB=4)
  {
    const int Mt = MROWS / 16;                   // 1024
    const int NtB = CH / 16 / 4;                 // 2 wave-columns of 4 tiles
    const int waves = Mt * NtB;                  // 2048
    wmma_gemm_out_kernel<4><<<(waves * 32 + TB - 1) / TB, TB, 0, stream>>>(
        y16, DI, wout16, DI, out, Mt, NtB, DI);
  }
}
